// Self_att_With_Adaptor_57062935495377
// MI455X (gfx1250) — compile-verified, matrix operand order checked by reading, not run on hardware
//
#include <hip/hip_runtime.h>
#include <hip/hip_bf16.h>

// ---------------------------------------------------------------------------
// Problem constants (from the reference)
// ---------------------------------------------------------------------------
#define S_TOK   32768
#define D_MODEL 1000
#define D_PAD   1008      // N padded to 63 tiles of 16
#define K_PAD   1024      // K padded to 32 tiles of 32
#define BN      100       // bottleneck
#define BN_PAD  128       // bottleneck padded (8 n-tiles / 4 k-tiles)

#define NT_D    63        // 1008 / 16
#define KT_D    32        // 1024 / 32
#define NT_B    8         // 128 / 16
#define KT_B    4         // 128 / 32

#define ROWS    32        // rows per block (two 16-row M tiles)
#define THREADS 256       // 8 waves of 32
#define WAVES   8

typedef __bf16 bf16_t;
typedef bf16_t v16bf __attribute__((ext_vector_type(16)));
typedef bf16_t v8bf  __attribute__((ext_vector_type(8)));
typedef float  v8f   __attribute__((ext_vector_type(8)));

union FragU { v16bf v; v8bf h[2]; };

// A-fragment (16x32 bf16, row-major source with stride `stride` elems).
// Lane L -> row (L&15); lane<16 covers K {0..7,16..23}, lane>=16 K {8..15,24..31}.
// So per lane: two contiguous 8-element (16B) chunks at koff and koff+16,
// where koff = kt*32 + hi*8.
static __device__ __forceinline__ v16bf load_afrag(const bf16_t* row_base, int koff) {
    FragU u;
    u.h[0] = *(const v8bf*)(row_base + koff);
    u.h[1] = *(const v8bf*)(row_base + koff + 16);
    return u.v;
}

// B-fragment from repacked weights: [nt][kt][lane][16] contiguous bf16.
static __device__ __forceinline__ v16bf load_bfrag(const bf16_t* wp, int kt_total,
                                                   int nt, int kt, int lane) {
    return *(const v16bf*)(wp + ((size_t)(nt * kt_total + kt) * 32 + lane) * 16);
}

static __device__ __forceinline__ v8f wmma_bf16(v16bf a, v16bf b, v8f c) {
    return __builtin_amdgcn_wmma_f32_16x16x32_bf16(false, a, false, b,
                                                   (short)0, c, false, false);
}

// ---------------------------------------------------------------------------
// Weight repack: f32 (K x N, row-major) -> bf16 fragment-native layout
// out[((nt*k_tiles + kt)*32 + lane)*16 + j] = bf16(W[k, n]) with zero padding
//   n = nt*16 + (lane&15); hi = lane>>4; k = kt*32 + (j<8 ? j : j+8) + hi*8
// ---------------------------------------------------------------------------
__global__ void repack_weight(const float* __restrict__ W, bf16_t* __restrict__ out,
                              int K, int N, int k_tiles, int n_tiles) {
    int total = n_tiles * k_tiles * 32 * 16;
    for (int idx = blockIdx.x * blockDim.x + threadIdx.x; idx < total;
         idx += gridDim.x * blockDim.x) {
        int j    = idx & 15;
        int lane = (idx >> 4) & 31;
        int t    = idx >> 9;
        int kt   = t % k_tiles;
        int nt   = t / k_tiles;
        int hi   = lane >> 4;
        int n    = nt * 16 + (lane & 15);
        int k    = kt * 32 + (j < 8 ? j : j + 8) + (hi ? 8 : 0);
        float v  = (k < K && n < N) ? W[(size_t)k * N + n] : 0.0f;
        out[idx] = (bf16_t)v;
    }
}

// ---------------------------------------------------------------------------
// Fused megakernel: LN -> v-proj -> o-proj(+residuals) -> adapter -> out
// One block = 32 token rows. All intermediates live in LDS.
// ---------------------------------------------------------------------------
__global__ __launch_bounds__(THREADS)
void fused_attn_adapter(const float* __restrict__ emb,
                        const float* __restrict__ ln_g,
                        const float* __restrict__ ln_b,
                        const float* __restrict__ bv,
                        const float* __restrict__ bo,
                        const float* __restrict__ bd,
                        const float* __restrict__ bu,
                        const bf16_t* __restrict__ Wv_p,
                        const bf16_t* __restrict__ Wo_p,
                        const bf16_t* __restrict__ Wd_p,
                        const bf16_t* __restrict__ Wu_p,
                        float* __restrict__ out) {
    extern __shared__ char smem[];
    float*  emb_s = (float*)smem;                            // 32*1000*4 = 128000 B (becomes emb1)
    bf16_t* x_s   = (bf16_t*)(smem + 128000);                // 32*1024*2 =  65536 B (x, then emb1 bf16)
    bf16_t* v_s   = (bf16_t*)(smem + 128000 + 65536);        // 32*1024*2 =  65536 B
    bf16_t* dn_s  = (bf16_t*)(smem + 128000 + 131072);       // 32*128*2  =   8192 B

    const int tid  = threadIdx.x;
    const int wave = tid >> 5;
    const int lane = tid & 31;
    const int hi   = lane >> 4;
    const int l16  = lane & 15;
    const int row0 = blockIdx.x * ROWS;

    // ---- zero K-pad regions (x_s cols 1000..1023, v_s cols 1008..1023) ----
    for (int i = tid; i < ROWS * (K_PAD - D_MODEL); i += THREADS) {
        int r = i / (K_PAD - D_MODEL), c = D_MODEL + i % (K_PAD - D_MODEL);
        x_s[r * K_PAD + c] = (bf16_t)0.0f;
    }
    for (int i = tid; i < ROWS * (K_PAD - D_PAD); i += THREADS) {
        int r = i / (K_PAD - D_PAD), c = D_PAD + i % (K_PAD - D_PAD);
        v_s[r * K_PAD + c] = (bf16_t)0.0f;
    }

    // ---- Phase 0: load emb tile + LayerNorm (8 threads per row) ----
    {
        const int q = tid & 7;
        const int r = tid >> 3;                 // 0..31
        const float* erow = emb + (size_t)(row0 + r) * D_MODEL;
        float s = 0.0f, ss = 0.0f;
        for (int c = q; c < D_MODEL; c += 8) {
            float e = erow[c];
            emb_s[r * D_MODEL + c] = e;
            s += e; ss += e * e;
        }
        #pragma unroll
        for (int m = 1; m < 8; m <<= 1) {
            s  += __shfl_xor(s,  m, 32);
            ss += __shfl_xor(ss, m, 32);
        }
        const float inv_n = 1.0f / (float)D_MODEL;
        float mu  = s * inv_n;
        float var = ss * inv_n - mu * mu;
        float rs  = rsqrtf(var + 1e-5f);
        for (int c = q; c < D_MODEL; c += 8) {
            float xv = (emb_s[r * D_MODEL + c] - mu) * rs * ln_g[c] + ln_b[c];
            x_s[r * K_PAD + c] = (bf16_t)xv;
        }
    }
    __syncthreads();

    // ---- Phase 1: v = x @ Wv + bv  (M=32, K=1024, N=1008) ----
    for (int nt = wave; nt < NT_D; nt += WAVES) {
        v8f acc0 = {}; v8f acc1 = {};
        for (int kt = 0; kt < KT_D; ++kt) {
            int koff = kt * 32 + hi * 8;
            v16bf b  = load_bfrag(Wv_p, KT_D, nt, kt, lane);
            v16bf a0 = load_afrag(x_s + (size_t)l16 * K_PAD, koff);
            v16bf a1 = load_afrag(x_s + (size_t)(16 + l16) * K_PAD, koff);
            acc0 = wmma_bf16(a0, b, acc0);
            acc1 = wmma_bf16(a1, b, acc1);
        }
        int col = nt * 16 + l16;
        float bias = (col < D_MODEL) ? bv[col] : 0.0f;
        #pragma unroll
        for (int r = 0; r < 8; ++r) {
            int mr = r + hi * 8;                       // row within tile
            v_s[(size_t)mr * K_PAD + col]        = (bf16_t)(acc0[r] + bias);
            v_s[(size_t)(16 + mr) * K_PAD + col] = (bf16_t)(acc1[r] + bias);
        }
    }
    __syncthreads();

    // ---- Phase 2: ao = v @ Wo + bo ; emb1 = emb + x + ao ----
    for (int nt = wave; nt < NT_D; nt += WAVES) {
        v8f acc0 = {}; v8f acc1 = {};
        for (int kt = 0; kt < KT_D; ++kt) {
            int koff = kt * 32 + hi * 8;
            v16bf b  = load_bfrag(Wo_p, KT_D, nt, kt, lane);
            v16bf a0 = load_afrag(v_s + (size_t)l16 * K_PAD, koff);
            v16bf a1 = load_afrag(v_s + (size_t)(16 + l16) * K_PAD, koff);
            acc0 = wmma_bf16(a0, b, acc0);
            acc1 = wmma_bf16(a1, b, acc1);
        }
        int col = nt * 16 + l16;
        float bias = (col < D_MODEL) ? bo[col] : 0.0f;
        #pragma unroll
        for (int r = 0; r < 8; ++r) {
            int mr = r + hi * 8;
            if (col < D_MODEL) {
                float e1a = emb_s[(size_t)mr * D_MODEL + col]
                          + (float)x_s[(size_t)mr * K_PAD + col] + acc0[r] + bias;
                emb_s[(size_t)mr * D_MODEL + col] = e1a;
                x_s[(size_t)mr * K_PAD + col]     = (bf16_t)e1a;
                float e1b = emb_s[(size_t)(16 + mr) * D_MODEL + col]
                          + (float)x_s[(size_t)(16 + mr) * K_PAD + col] + acc1[r] + bias;
                emb_s[(size_t)(16 + mr) * D_MODEL + col] = e1b;
                x_s[(size_t)(16 + mr) * K_PAD + col]     = (bf16_t)e1b;
            } else {
                x_s[(size_t)mr * K_PAD + col]        = (bf16_t)0.0f;
                x_s[(size_t)(16 + mr) * K_PAD + col] = (bf16_t)0.0f;
            }
        }
    }
    __syncthreads();

    // ---- Phase 3: down = relu(emb1 @ Wd + bd)  (N=128, one n-tile per wave) ----
    {
        int nt = wave;                                 // 8 waves, 8 tiles
        v8f acc0 = {}; v8f acc1 = {};
        for (int kt = 0; kt < KT_D; ++kt) {
            int koff = kt * 32 + hi * 8;
            v16bf b  = load_bfrag(Wd_p, KT_D, nt, kt, lane);
            v16bf a0 = load_afrag(x_s + (size_t)l16 * K_PAD, koff);
            v16bf a1 = load_afrag(x_s + (size_t)(16 + l16) * K_PAD, koff);
            acc0 = wmma_bf16(a0, b, acc0);
            acc1 = wmma_bf16(a1, b, acc1);
        }
        int col = nt * 16 + l16;
        float bias = (col < BN) ? bd[col] : 0.0f;
        #pragma unroll
        for (int r = 0; r < 8; ++r) {
            int mr = r + hi * 8;
            float d0 = acc0[r] + bias; d0 = d0 > 0.0f ? d0 : 0.0f;
            float d1 = acc1[r] + bias; d1 = d1 > 0.0f ? d1 : 0.0f;
            dn_s[(size_t)mr * BN_PAD + col]        = (bf16_t)d0;
            dn_s[(size_t)(16 + mr) * BN_PAD + col] = (bf16_t)d1;
        }
    }
    __syncthreads();

    // ---- Phase 4: out = 2*emb1 + down @ Wu + bu ----
    for (int nt = wave; nt < NT_D; nt += WAVES) {
        v8f acc0 = {}; v8f acc1 = {};
        for (int kt = 0; kt < KT_B; ++kt) {
            int koff = kt * 32 + hi * 8;
            v16bf b  = load_bfrag(Wu_p, KT_B, nt, kt, lane);
            v16bf a0 = load_afrag(dn_s + (size_t)l16 * BN_PAD, koff);
            v16bf a1 = load_afrag(dn_s + (size_t)(16 + l16) * BN_PAD, koff);
            acc0 = wmma_bf16(a0, b, acc0);
            acc1 = wmma_bf16(a1, b, acc1);
        }
        int col = nt * 16 + l16;
        if (col < D_MODEL) {
            float bias = bu[col];
            #pragma unroll
            for (int r = 0; r < 8; ++r) {
                int mr = r + hi * 8;
                out[(size_t)(row0 + mr) * D_MODEL + col] =
                    2.0f * emb_s[(size_t)mr * D_MODEL + col] + acc0[r] + bias;
                out[(size_t)(row0 + 16 + mr) * D_MODEL + col] =
                    2.0f * emb_s[(size_t)(16 + mr) * D_MODEL + col] + acc1[r] + bias;
            }
        }
    }
}

// ---------------------------------------------------------------------------
// Host-side launch
// Input order: emb(0) ln_g(1) ln_b(2) Wq(3) bq(4) Wk(5) bk(6)
//              Wv(7) bv(8) Wo(9) bo(10) Wd(11) bd(12) Wu(13) bu(14)
// ---------------------------------------------------------------------------
extern "C" void kernel_launch(void* const* d_in, const int* in_sizes, int n_in,
                              void* d_out, int out_size, void* d_ws, size_t ws_size,
                              hipStream_t stream) {
    const float* emb  = (const float*)d_in[0];
    const float* ln_g = (const float*)d_in[1];
    const float* ln_b = (const float*)d_in[2];
    const float* Wv   = (const float*)d_in[7];
    const float* bv   = (const float*)d_in[8];
    const float* Wo   = (const float*)d_in[9];
    const float* bo   = (const float*)d_in[10];
    const float* Wd   = (const float*)d_in[11];
    const float* bd   = (const float*)d_in[12];
    const float* Wu   = (const float*)d_in[13];
    const float* bu   = (const float*)d_in[14];
    float* out = (float*)d_out;

    // Workspace layout (bf16 elements)
    const size_t WV_ELEMS = (size_t)NT_D * KT_D * 512;  // 1,032,192
    const size_t WD_ELEMS = (size_t)NT_B * KT_D * 512;  //   131,072
    const size_t WU_ELEMS = (size_t)NT_D * KT_B * 512;  //   129,024
    bf16_t* Wv_p = (bf16_t*)d_ws;
    bf16_t* Wo_p = Wv_p + WV_ELEMS;
    bf16_t* Wd_p = Wo_p + WV_ELEMS;
    bf16_t* Wu_p = Wd_p + WD_ELEMS;
    (void)Wu_p; (void)ws_size; (void)n_in; (void)in_sizes; (void)out_size;

    // Repack weights into fragment-native bf16 layout (zero-padded)
    repack_weight<<<1024, 256, 0, stream>>>(Wv, Wv_p, D_MODEL, D_MODEL, KT_D, NT_D);
    repack_weight<<<1024, 256, 0, stream>>>(Wo, Wo_p, D_MODEL, D_MODEL, KT_D, NT_D);
    repack_weight<<<256,  256, 0, stream>>>(Wd, Wd_p, D_MODEL, BN,      KT_D, NT_B);
    repack_weight<<<256,  256, 0, stream>>>(Wu, Wu_p, BN,      D_MODEL, KT_B, NT_D);

    // Fused megakernel: 1024 blocks x 256 threads, 261 KB dynamic LDS
    const size_t smem_bytes = 128000 + 65536 + 65536 + 8192;  // 267,264 B
    dim3 grid(S_TOK / ROWS);
    dim3 block(THREADS);
    fused_attn_adapter<<<grid, block, smem_bytes, stream>>>(
        emb, ln_g, ln_b, bv, bo, bd, bu, Wv_p, Wo_p, Wd_p, Wu_p, out);
}
